// PATEncoder_1400159338792
// MI455X (gfx1250) — matrix operand order repaired and checked by reading, not compiled, not run on hardware
//
#include <hip/hip_runtime.h>
#include <hip/hip_fp16.h>
#include <math.h>

typedef _Float16 f16;
typedef __attribute__((ext_vector_type(16))) _Float16 v16h;
typedef __attribute__((ext_vector_type(8)))  _Float16 v8h;
typedef __attribute__((ext_vector_type(8)))  float    v8f;
typedef __attribute__((ext_vector_type(4)))  unsigned int v4u;
typedef __attribute__((ext_vector_type(4)))  int v4i;
typedef __attribute__((ext_vector_type(8)))  int v8i;

constexpr int B_ = 64;
constexpr int S_ = 560;
constexpr int E_ = 96;
constexpr int H_ = 12;
constexpr int D_ = 96;
constexpr int F_ = 256;
constexpr int PATCH_ = 18;
constexpr int M_ = B_ * S_;        // 35840 token rows
constexpr int TPAD = 576;          // S padded to multiple of 32 (WMMA K-dim)
constexpr int KT_LD = 592;         // padded LDS leading dim for K^T (bank spread)
constexpr float SCALE_ = 0.10206207261596575f; // 1/sqrt(96)

// LDS layout of the attention workgroup (dynamic LDS starts at offset 0)
constexpr unsigned LDS_KT_OFF = 0;
constexpr unsigned LDS_VS_OFF = (unsigned)(E_ * KT_LD * 2);   // 113664
constexpr unsigned LDS_SC_OFF = LDS_VS_OFF + (unsigned)(TPAD * D_ * 2);
constexpr unsigned LDS_PS_OFF = LDS_SC_OFF + (unsigned)(16 * TPAD * 4);

// ---------------- WMMA helpers (CDNA5 16x16x32 f16 -> f32) ----------------

__device__ __forceinline__ v8f wmma_acc(v16h a, v16h b, v8f c) {
  // (neg_a, A, neg_b, B, c_mod, C, reuse_a, reuse_b)
  return __builtin_amdgcn_wmma_f32_16x16x32_f16(false, a, false, b, (short)0, c,
                                                false, false);
}

// A fragment 16x32 f16: lane L<16 -> row L, K = {0..7,16..23}; lane L>=16 -> row L-16,
// K = {8..15,24..31}. rowptr already points at this lane's row.
__device__ __forceinline__ v16h fragA(const f16* rowptr, int k0, int lane) {
  const int off = (lane & 16) ? 8 : 0;
  v8h lo = *(const v8h*)(rowptr + k0 + off);
  v8h hi = *(const v8h*)(rowptr + k0 + 16 + off);
  return __builtin_shufflevector(lo, hi, 0,1,2,3,4,5,6,7,8,9,10,11,12,13,14,15);
}

// B fragment 32x16 f16: lane L -> column L%16, K = {0..7,16..23} (L<16) or {8..15,24..31}.
// p = &W[k0*ld + n0]
__device__ __forceinline__ v16h fragB(const f16* p, int ld, int lane) {
  const int off = (lane & 16) ? 8 : 0;
  const int n = lane & 15;
  v16h r;
#pragma unroll
  for (int s = 0; s < 8; ++s) {
    r[s] = p[(size_t)(off + s) * ld + n];
    r[8 + s] = p[(size_t)(16 + off + s) * ld + n];
  }
  return r;
}

// ---------------- small utility kernels ----------------

__global__ void cvt_f32_f16(const float* __restrict__ src, f16* __restrict__ dst, int n) {
  for (int i = blockIdx.x * blockDim.x + threadIdx.x; i < n; i += gridDim.x * blockDim.x)
    dst[i] = (f16)src[i];
}

__global__ void patch_embed(const float* __restrict__ x, const float* __restrict__ pw,
                            const float* __restrict__ pb,
                            float* __restrict__ h32, f16* __restrict__ h16) {
  int idx = blockIdx.x * blockDim.x + threadIdx.x;   // over M_*E_
  if (idx >= M_ * E_) return;
  int e = idx % E_;
  int row = idx / E_;
  int s = row % S_;
  int b = row / S_;
  const float* xp = x + (size_t)b * (S_ * PATCH_) + s * PATCH_;
  float acc = pb[e];
#pragma unroll
  for (int p = 0; p < PATCH_; ++p) acc += xp[p] * pw[p * E_ + e];
  // sinusoidal PE
  float freq = __expf(-(float)(e & ~1) * (9.210340371976184f / (float)E_));
  float ang = (float)s * freq;
  acc += (e & 1) ? __cosf(ang) : __sinf(ang);
  h32[idx] = acc;
  h16[idx] = (f16)acc;
}

// residual add + layernorm; one wave per 96-wide row, 8 rows per block
__global__ void add_ln(const float* __restrict__ hin, const float* __restrict__ res,
                       const float* __restrict__ g, const float* __restrict__ bt,
                       float* __restrict__ hout, f16* __restrict__ h16out) {
  const int lane = threadIdx.x & 31;
  const int wave = threadIdx.x >> 5;
  const int row = blockIdx.x * 8 + wave;
  const size_t base = (size_t)row * E_;
  float vv[3];
#pragma unroll
  for (int j = 0; j < 3; ++j) {
    int e = lane + 32 * j;
    vv[j] = hin[base + e] + res[base + e];
  }
  float s1 = vv[0] + vv[1] + vv[2];
  float s2 = vv[0]*vv[0] + vv[1]*vv[1] + vv[2]*vv[2];
#pragma unroll
  for (int o = 16; o; o >>= 1) {
    s1 += __shfl_xor(s1, o, 32);
    s2 += __shfl_xor(s2, o, 32);
  }
  float mean = s1 * (1.f / 96.f);
  float var  = s2 * (1.f / 96.f) - mean * mean;
  float rstd = rsqrtf(var + 1e-5f);
#pragma unroll
  for (int j = 0; j < 3; ++j) {
    int e = lane + 32 * j;
    float y = (vv[j] - mean) * rstd * g[e] + bt[e];
    hout[base + e] = y;
    h16out[base + e] = (f16)y;
  }
}

// ---------------- generic WMMA GEMM: C[M,N] = A[M,K](f16) * W[K,N](f16) + bias ----------
// One wave computes one 16x16 tile. grid = (M/16, ceil(N/64), Z); 128 threads (4 waves).
// Z indexes heads via the stride parameters.

__global__ void gemm_ws(const f16* __restrict__ A, const f16* __restrict__ W,
                        const float* __restrict__ bias,
                        float* __restrict__ out32, f16* __restrict__ out16,
                        int N, int K,
                        long long strideW, long long strideB, long long strideOut,
                        int relu) {
  const int lane = threadIdx.x & 31;
  const int wave = threadIdx.x >> 5;
  const int z = blockIdx.z;
  W    += (size_t)z * strideW;
  bias += (size_t)z * strideB;
  const int n0 = (blockIdx.y * 4 + wave) * 16;
  if (n0 >= N) return;
  const int row0 = blockIdx.x * 16;
  const int rl = lane & 15;
  const f16* arow = A + (size_t)(row0 + rl) * K;
  v8f acc = {0,0,0,0,0,0,0,0};
  for (int k0 = 0; k0 < K; k0 += 32) {
    v16h a = fragA(arow, k0, lane);
    v16h b = fragB(W + (size_t)k0 * N + n0, N, lane);
    acc = wmma_acc(a, b, acc);
  }
  const int col = lane & 15;           // D frag: lane = N
  const int rh  = (lane >> 4) * 8;     // D frag: vgpr c -> row c (+8 for upper lanes)
  const float bv = bias[n0 + col];
#pragma unroll
  for (int c = 0; c < 8; ++c) {
    float v = acc[c] + bv;
    if (relu) v = fmaxf(v, 0.f);
    size_t o = (size_t)z * strideOut + (size_t)(row0 + rh + c) * N + n0 + col;
    if (out32) out32[o] = v;
    if (out16) out16[o] = (f16)v;
  }
}

// ---------------- fused attention: one workgroup per (b,h), all of K,V in LDS --------
// q/k/v layout: [H][B*S][D] f16. Output written into concat [B*S][H*D] f16.
// V is staged by the Tensor Data Mover (async DMA -> LDS, TENSORcnt); K is staged
// transposed by the waves (vector global loads, scalar LDS scatter).

__global__ void attention(const f16* __restrict__ q, const f16* __restrict__ k,
                          const f16* __restrict__ v, f16* __restrict__ concat) {
  extern __shared__ char smem[];
  f16*   Kt = (f16*)smem;                    // [D_][KT_LD]   K transposed (d-major)
  f16*   Vs = Kt + E_ * KT_LD;               // [TPAD][D_]    V row-major, padded rows zero
  float* Sc = (float*)(Vs + TPAD * D_);      // [16][TPAD]    fp32 score strip
  f16*   Ps = (f16*)(Sc + 16 * TPAD);        // [16][TPAD]    f16 softmax probs

  const int tid  = threadIdx.x;
  const int lane = tid & 31;
  const int wave = tid >> 5;
  const int h = blockIdx.x % H_;
  const int b = blockIdx.x / H_;
  const size_t headOff = ((size_t)h * M_ + (size_t)b * S_) * D_;
  const f16* kb = k + headOff;
  const f16* vb = v + headOff;
  const f16* qb = q + headOff;

  // ---- async DMA: V tile (560*96 f16 = 13440 8-byte units) -> LDS via TDM ----
  if (wave == 0) {
    unsigned long long gaddr = (unsigned long long)(const void*)vb;
    constexpr int n8 = S_ * D_ * 2 / 8;            // 13440 elements of 8 bytes
    v4u g0;
    g0.x = 1u;                                      // count=1, user descriptor
    g0.y = LDS_VS_OFF;                              // lds_addr (bytes)
    g0.z = (unsigned)(gaddr & 0xffffffffull);       // global_addr[31:0]
    g0.w = (unsigned)((gaddr >> 32) & 0x01ffffffull) | (2u << 30); // addr[56:32] | type=2
    v8i g1;
    g1[0] = (3 << 16);                              // workgroup_mask=0, data_size=3 (8B)
    g1[1] = (n8 & 0xffff) << 16;                    // tensor_dim0[15:0]
    g1[2] = (n8 >> 16) | (1 << 16);                 // tensor_dim0[31:16] | tensor_dim1=1
    g1[3] = (n8 & 0xffff) << 16;                    // tile_dim0 = n8
    g1[4] = 0;                                      // tile_dim1/2 unused
    g1[5] = n8;                                     // tensor_dim0_stride
    g1[6] = 0;
    g1[7] = 0;
    v4i gz = {0, 0, 0, 0};
#if defined(__clang_major__) && (__clang_major__ >= 23)
    v8i gz8 = {0, 0, 0, 0, 0, 0, 0, 0};
    __builtin_amdgcn_tensor_load_to_lds(g0, g1, gz, gz, gz8, 0);
#else
    __builtin_amdgcn_tensor_load_to_lds(g0, g1, gz, gz, 0);
#endif
  }

  // ---- meanwhile: stage K transposed (vector global loads, strided LDS scatter) ----
  constexpr int DB = D_ / 8;                        // 12 8-element chunks per row
  for (int i = tid; i < S_ * DB; i += blockDim.x) {
    int t = i / DB, db = i % DB;
    v8h kv = *(const v8h*)(kb + (size_t)t * D_ + db * 8);
#pragma unroll
    for (int j = 0; j < 8; ++j) Kt[(db * 8 + j) * KT_LD + t] = kv[j];
  }
  // zero V pad rows (t in [560,576)) -- disjoint from the TDM destination
  {
    v8h z8 = {0,0,0,0,0,0,0,0};
    v8h* vpad = (v8h*)(Vs + S_ * D_);
    for (int i = tid; i < (TPAD - S_) * D_ / 8; i += blockDim.x) vpad[i] = z8;
  }
  if (wave == 0) __builtin_amdgcn_s_wait_tensorcnt(0);
  __syncthreads();

  const int rl  = lane & 15;
  const int col = lane & 15;
  const int rh  = (lane >> 4) * 8;

  for (int rb = 0; rb < S_ / 16; ++rb) {
    // prefetch next Q row-block while computing this one
    if (rb + 1 < S_ / 16)
      __builtin_prefetch(qb + (size_t)((rb + 1) * 16 + rl) * D_, 0, 1);

    // ---- scores: Sc = scale * Q_blk * K^T  (35 column tiles spread over 8 waves) ----
    const f16* qrow = qb + (size_t)(rb * 16 + rl) * D_;
    for (int ct = wave; ct < S_ / 16; ct += 8) {
      v8f acc = {0,0,0,0,0,0,0,0};
#pragma unroll
      for (int ks = 0; ks < D_ / 32; ++ks) {
        v16h a  = fragA(qrow, ks * 32, lane);
        v16h bf = fragB(Kt + (size_t)(ks * 32) * KT_LD + ct * 16, KT_LD, lane);
        acc = wmma_acc(a, bf, acc);
      }
#pragma unroll
      for (int c = 0; c < 8; ++c)
        Sc[(rh + c) * TPAD + ct * 16 + col] = acc[c] * SCALE_;
    }
    __syncthreads();

    // ---- softmax over 560 cols: 16 threads per row, shfl reductions ----
    {
      const int r  = tid >> 4;
      const int sb = tid & 15;
      float m = -3.0e38f;
      for (int c = sb; c < S_; c += 16) m = fmaxf(m, Sc[r * TPAD + c]);
      for (int o = 8; o; o >>= 1) m = fmaxf(m, __shfl_xor(m, o, 16));
      float sum = 0.f;
      for (int c = sb; c < S_; c += 16) {
        float ev = __expf(Sc[r * TPAD + c] - m);
        Sc[r * TPAD + c] = ev;
        sum += ev;
      }
      for (int o = 8; o; o >>= 1) sum += __shfl_xor(sum, o, 16);
      float inv = 1.f / sum;
      for (int c = sb; c < S_; c += 16)
        Ps[r * TPAD + c] = (f16)(Sc[r * TPAD + c] * inv);
      Ps[r * TPAD + S_ + sb] = (f16)0;   // zero padded K-dim columns
    }
    __syncthreads();

    // ---- O_blk = P * V : six 16-wide d-tiles, waves 0..5 ----
    if (wave < D_ / 16) {
      v8f acc = {0,0,0,0,0,0,0,0};
      for (int ks = 0; ks < TPAD / 32; ++ks) {
        v16h a  = fragA(Ps + (size_t)rl * TPAD, ks * 32, lane);
        v16h bf = fragB(Vs + (size_t)(ks * 32) * D_ + wave * 16, D_, lane);
        acc = wmma_acc(a, bf, acc);
      }
      size_t base = ((size_t)b * S_ + rb * 16) * (H_ * D_) + h * D_ + wave * 16 + col;
#pragma unroll
      for (int c = 0; c < 8; ++c)
        concat[base + (size_t)(rh + c) * (H_ * D_)] = (f16)acc[c];
    }
    __syncthreads();
  }
}

// ---------------- host orchestration ----------------

extern "C" void kernel_launch(void* const* d_in, const int* in_sizes, int n_in,
                              void* d_out, int out_size, void* d_ws, size_t ws_size,
                              hipStream_t stream) {
  (void)in_sizes; (void)n_in; (void)out_size; (void)ws_size;
  const float* x   = (const float*)d_in[0];
  const float* pW  = (const float*)d_in[1];
  const float* pb  = (const float*)d_in[2];
  const float* Wq  = (const float*)d_in[3];
  const float* bq  = (const float*)d_in[4];
  const float* Wk  = (const float*)d_in[5];
  const float* bk  = (const float*)d_in[6];
  const float* Wv  = (const float*)d_in[7];
  const float* bv  = (const float*)d_in[8];
  const float* Wo  = (const float*)d_in[9];
  const float* bo  = (const float*)d_in[10];
  const float* f1W = (const float*)d_in[11];
  const float* f1b = (const float*)d_in[12];
  const float* f2W = (const float*)d_in[13];
  const float* f2b = (const float*)d_in[14];
  const float* n1g = (const float*)d_in[15];
  const float* n1b = (const float*)d_in[16];
  const float* n2g = (const float*)d_in[17];
  const float* n2b = (const float*)d_in[18];
  float* out = (float*)d_out;

  char* p = (char*)d_ws;
  auto take = [&](size_t bytes) { char* r = p; p += (bytes + 255) & ~(size_t)255; return r; };
  float* h32   = (float*)take((size_t)M_ * E_ * 4);
  float* tmp32 = (float*)take((size_t)M_ * E_ * 4);
  f16*   h16   = (f16*)take((size_t)M_ * E_ * 2);
  f16*   q16   = (f16*)take((size_t)H_ * M_ * D_ * 2);
  f16*   k16   = (f16*)take((size_t)H_ * M_ * D_ * 2);
  f16*   v16   = (f16*)take((size_t)H_ * M_ * D_ * 2);
  f16*   cat16 = (f16*)take((size_t)M_ * H_ * D_ * 2);
  f16*   ff16  = (f16*)take((size_t)M_ * F_ * 2);
  f16*   wq16  = (f16*)take((size_t)2 * H_ * E_ * D_ * 2);
  f16*   wk16  = (f16*)take((size_t)2 * H_ * E_ * D_ * 2);
  f16*   wv16  = (f16*)take((size_t)2 * H_ * E_ * D_ * 2);
  f16*   wo16  = (f16*)take((size_t)2 * (H_ * D_) * E_ * 2);
  f16*   w116  = (f16*)take((size_t)2 * E_ * F_ * 2);
  f16*   w216  = (f16*)take((size_t)2 * F_ * E_ * 2);

  auto cvt = [&](const float* s, f16* d, int n) {
    cvt_f32_f16<<<(n + 255) / 256, 256, 0, stream>>>(s, d, n);
  };
  cvt(Wq, wq16, 2 * H_ * E_ * D_);
  cvt(Wk, wk16, 2 * H_ * E_ * D_);
  cvt(Wv, wv16, 2 * H_ * E_ * D_);
  cvt(Wo, wo16, 2 * (H_ * D_) * E_);
  cvt(f1W, w116, 2 * E_ * F_);
  cvt(f2W, w216, 2 * F_ * E_);

  patch_embed<<<(M_ * E_ + 255) / 256, 256, 0, stream>>>(x, pW, pb, h32, h16);

  constexpr int ATTN_LDS = E_ * KT_LD * 2 + TPAD * D_ * 2 + 16 * TPAD * 4 + 16 * TPAD * 2;
  (void)hipFuncSetAttribute(reinterpret_cast<const void*>(attention),
                            hipFuncAttributeMaxDynamicSharedMemorySize, ATTN_LDS);

  for (int l = 0; l < 2; ++l) {
    const long long wOff = (long long)l * H_ * E_ * D_;
    // QKV projections: grid.z = heads
    dim3 gq(M_ / 16, (E_ + 63) / 64, H_);
    gemm_ws<<<gq, 128, 0, stream>>>(h16, wq16 + wOff, bq + l * H_ * D_, nullptr, q16,
                                    D_, E_, (long long)E_ * D_, D_, (long long)M_ * D_, 0);
    gemm_ws<<<gq, 128, 0, stream>>>(h16, wk16 + wOff, bk + l * H_ * D_, nullptr, k16,
                                    D_, E_, (long long)E_ * D_, D_, (long long)M_ * D_, 0);
    gemm_ws<<<gq, 128, 0, stream>>>(h16, wv16 + wOff, bv + l * H_ * D_, nullptr, v16,
                                    D_, E_, (long long)E_ * D_, D_, (long long)M_ * D_, 0);

    attention<<<B_ * H_, 256, ATTN_LDS, stream>>>(q16, k16, v16, cat16);

    dim3 go(M_ / 16, (E_ + 63) / 64, 1);
    gemm_ws<<<go, 128, 0, stream>>>(cat16, wo16 + (long long)l * H_ * D_ * E_, bo + l * E_,
                                    tmp32, nullptr, E_, H_ * D_, 0, 0, 0, 0);
    add_ln<<<M_ / 8, 256, 0, stream>>>(h32, tmp32, n1g + l * E_, n1b + l * E_, h32, h16);

    dim3 g1(M_ / 16, (F_ + 63) / 64, 1);
    gemm_ws<<<g1, 128, 0, stream>>>(h16, w116 + (long long)l * E_ * F_, f1b + l * F_,
                                    nullptr, ff16, F_, E_, 0, 0, 0, 1);
    dim3 g2(M_ / 16, (E_ + 63) / 64, 1);
    gemm_ws<<<g2, 128, 0, stream>>>(ff16, w216 + (long long)l * F_ * E_, f2b + l * E_,
                                    tmp32, nullptr, E_, F_, 0, 0, 0, 0);

    float* lnout = (l == 1) ? out : h32;
    add_ln<<<M_ / 8, 256, 0, stream>>>(h32, tmp32, n2g + l * E_, n2b + l * E_, lnout, h16);
  }
}
